// MVP_CrossAttn_EnvGrid_64759516889863
// MI455X (gfx1250) — matrix operand order repaired and checked by reading, not compiled, not run on hardware
//
#include <hip/hip_runtime.h>

typedef __attribute__((ext_vector_type(16))) _Float16 v16h;
typedef __attribute__((ext_vector_type(8)))  _Float16 v8h;
typedef __attribute__((ext_vector_type(8)))  float    v8f;

constexpr int kD   = 256;
constexpr int kNH  = 8;
constexpr int kDH  = 32;
constexpr int kG   = 64;
constexpr int kK   = 4096;   // kG*kG
constexpr int kB   = 4;
constexpr int kN   = 512;
constexpr int kNs  = 513;    // N + 1 (bs token)
constexpr int kNsP = 544;    // padded to multiple of 32
constexpr int kDFF = 1024;
constexpr int kEC  = 32;
constexpr int kCity= 256;
constexpr int kT   = 4;

__device__ __forceinline__ float gelu_f(float x) {
  float x3 = x * x * x;
  return 0.5f * x * (1.0f + tanhf(0.7978845608028654f * (x + 0.044715f * x3)));
}

// ---------------------------------------------------------------------------
// f32 [nb][Kd][N] -> f16 transposed [nb][N][Kd]  (weights, so B-fragments are
// contiguous per-lane b128 loads in the WMMA GEMM)
// ---------------------------------------------------------------------------
__global__ void cvt_transpose_f16(const float* __restrict__ src, _Float16* __restrict__ dst,
                                  int Kd, int N, int nb) {
  long idx = (long)blockIdx.x * blockDim.x + threadIdx.x;
  long tot = (long)nb * Kd * N;
  if (idx >= tot) return;
  int n  = (int)(idx % N);
  long r = idx / N;
  int k  = (int)(r % Kd);
  int bi = (int)(r / Kd);
  dst[(long)bi * Kd * N + (long)n * Kd + k] = (_Float16)src[idx];
}

// ---------------------------------------------------------------------------
// Token MLP layer 1 (reduction of 3) -> f16, rows >= N zeroed (padding + bs slot)
// ---------------------------------------------------------------------------
__global__ void h1_kernel(const float* __restrict__ meas_xy, const float* __restrict__ meas_v,
                          const float* __restrict__ w1, const float* __restrict__ b1,
                          _Float16* __restrict__ h1) {
  long idx = (long)blockIdx.x * blockDim.x + threadIdx.x;
  if (idx >= (long)kB * kNsP * kD) return;
  int d   = (int)(idx % kD);
  long r  = idx / kD;
  int n   = (int)(r % kNsP);
  int b   = (int)(r / kNsP);
  float val = 0.0f;
  if (n < kN) {
    float x = meas_xy[((long)b * kN + n) * 2 + 0];
    float y = meas_xy[((long)b * kN + n) * 2 + 1];
    float v = meas_v[(long)b * kN + n];
    val = gelu_f(x * w1[d] + y * w1[kD + d] + v * w1[2 * kD + d] + b1[d]);
  }
  h1[idx] = (_Float16)val;
}

// ---------------------------------------------------------------------------
// bs token (full 2-layer MLP, one block per batch), writes src16 row N
// ---------------------------------------------------------------------------
__global__ __launch_bounds__(256) void bs_kernel(const float* __restrict__ bs_xy,
                          const float* __restrict__ w1, const float* __restrict__ b1,
                          const float* __restrict__ w2, const float* __restrict__ b2,
                          _Float16* __restrict__ src16) {
  __shared__ float u[kD];
  int b = blockIdx.x, d = threadIdx.x;
  float x = bs_xy[b * 2 + 0], y = bs_xy[b * 2 + 1];
  u[d] = gelu_f(x * w1[d] + y * w1[kD + d] + b1[d]);
  __syncthreads();
  float s = b2[d];
  for (int j = 0; j < kD; j++) s += u[j] * w2[j * kD + d];
  src16[((long)b * kNsP + kN) * kD + d] = (_Float16)s;
}

// ---------------------------------------------------------------------------
// task_vec[b,d] = task_emb[task_id[b]] @ tp_w + tp_b
// ---------------------------------------------------------------------------
__global__ __launch_bounds__(256) void taskvec_kernel(const int* __restrict__ task_id,
                          const float* __restrict__ emb, const float* __restrict__ tw,
                          const float* __restrict__ tb, float* __restrict__ tv) {
  int b = blockIdx.x, d = threadIdx.x;
  int t = task_id[b];
  float s = tb[d];
  for (int i = 0; i < kD; i++) s += emb[t * kD + i] * tw[i * kD + d];
  tv[b * kD + d] = s;
}

// ---------------------------------------------------------------------------
// Gaussian measurement density, meas staged in LDS
// ---------------------------------------------------------------------------
__global__ __launch_bounds__(256) void density_kernel(const float* __restrict__ meas_xy,
                                                      float* __restrict__ dens) {
  __shared__ float mx[kN], my[kN];
  int b = blockIdx.y, t = threadIdx.x;
  for (int i = t; i < kN; i += 256) {
    mx[i] = meas_xy[((long)b * kN + i) * 2 + 0];
    my[i] = meas_xy[((long)b * kN + i) * 2 + 1];
  }
  __syncthreads();
  int k = blockIdx.x * 256 + t;
  float gx = ((k & 63) + 0.5f) * (1.0f / kG);
  float gy = ((k >> 6) + 0.5f) * (1.0f / kG);
  float s = 0.0f;
  for (int i = 0; i < kN; i++) {
    float dx = gx - mx[i], dy = gy - my[i];
    s += __expf(-(dx * dx + dy * dy) * 78.125f);   // 1/(2*0.08^2)
  }
  dens[(long)b * kK + k] = s * (1.0f / kN);
}

// ---------------------------------------------------------------------------
// Env CNN
// ---------------------------------------------------------------------------
__global__ void conv1_k(const float* __restrict__ city, const float* __restrict__ w,
                        const float* __restrict__ bias, float* __restrict__ out) {
  int idx = blockIdx.x * blockDim.x + threadIdx.x;
  if (idx >= kB * kEC * kG * kG) return;
  int x = idx & 63, y = (idx >> 6) & 63, c = (idx >> 12) & 31, b = idx >> 17;
  float s = bias[c];
  for (int dy = 0; dy < 3; dy++)
    for (int dx = 0; dx < 3; dx++) {
      int yy = y + dy - 1, xx = x + dx - 1;
      if (yy >= 0 && yy < kG && xx >= 0 && xx < kG)
        s += city[(long)b * kCity * kCity + (yy * 4) * kCity + xx * 4] * w[c * 9 + dy * 3 + dx];
    }
  out[idx] = gelu_f(s);
}

__global__ void conv2_k(const float* __restrict__ in, const float* __restrict__ w,
                        const float* __restrict__ bias, float* __restrict__ out) {
  int idx = blockIdx.x * blockDim.x + threadIdx.x;
  if (idx >= kB * kEC * kG * kG) return;
  int x = idx & 63, y = (idx >> 6) & 63, c = (idx >> 12) & 31, b = idx >> 17;
  float s = bias[c];
  for (int ci = 0; ci < kEC; ci++)
    for (int dy = 0; dy < 3; dy++)
      for (int dx = 0; dx < 3; dx++) {
        int yy = y + dy - 1, xx = x + dx - 1;
        if (yy >= 0 && yy < kG && xx >= 0 && xx < kG)
          s += in[((long)(b * kEC + ci) << 12) + (yy << 6) + xx] * w[((c * kEC + ci) * 3 + dy) * 3 + dx];
      }
  out[idx] = gelu_f(s);
}

// 1x1 conv + grid_pos + task_vec -> grid latent
__global__ void conv3_assemble(const float* __restrict__ c2, const float* __restrict__ w3,
                               const float* __restrict__ b3, const float* __restrict__ grid_pos,
                               const float* __restrict__ tv, float* __restrict__ gr) {
  long idx = (long)blockIdx.x * blockDim.x + threadIdx.x;
  if (idx >= (long)kB * kK * kD) return;
  int d = (int)(idx % kD);
  long r = idx / kD;
  int k = (int)(r % kK), b = (int)(r / kK);
  int y = k >> 6, x = k & 63;
  float s = b3[d];
  const float* cc = c2 + (long)b * kEC * kG * kG + y * kG + x;
  for (int ci = 0; ci < kEC; ci++) s += cc[ci * kG * kG] * w3[d * kEC + ci];
  gr[idx] = s + grid_pos[k * kD + d] + tv[b * kD + d];
}

// ---------------------------------------------------------------------------
// density modulation: out = in + density*dens_w + dens_b
// ---------------------------------------------------------------------------
__global__ void densadd_k(const float* __restrict__ in, const float* __restrict__ dens,
                          const float* __restrict__ dw, const float* __restrict__ db,
                          float* __restrict__ out) {
  long idx = (long)blockIdx.x * blockDim.x + threadIdx.x;
  if (idx >= (long)kB * kK * kD) return;
  int d = (int)(idx % kD);
  long row = idx / kD;
  out[idx] = in[idx] + dens[row] * dw[d] + db[d];
}

// ---------------------------------------------------------------------------
// LayerNorm over D=256, one block per row, f16 output
// ---------------------------------------------------------------------------
__global__ __launch_bounds__(256) void ln_kernel(const float* __restrict__ in,
                          _Float16* __restrict__ out, const float* __restrict__ w,
                          const float* __restrict__ b) {
  __shared__ float red[256];
  long row = blockIdx.x;
  int t = threadIdx.x;
  float v = in[row * kD + t];
  red[t] = v; __syncthreads();
  for (int s = 128; s > 0; s >>= 1) { if (t < s) red[t] += red[t + s]; __syncthreads(); }
  float mean = red[0] * (1.0f / kD);
  __syncthreads();
  float d = v - mean;
  red[t] = d * d; __syncthreads();
  for (int s = 128; s > 0; s >>= 1) { if (t < s) red[t] += red[t + s]; __syncthreads(); }
  float var = red[0] * (1.0f / kD);
  out[row * kD + t] = (_Float16)(d * rsqrtf(var + 1e-5f) * w[t] + b[t]);
}

// ---------------------------------------------------------------------------
// WMMA GEMM: C = act(A[M,Kd] @ W[Kd,N] + bias (+ residual)); W stored transposed
// f16 [N,Kd]. One wave per 16x64 tile (4 accumulators): the A fragment is
// loaded once per k-step and reused across 4 v_wmma_f32_16x16x32_f16, raising
// arithmetic intensity from ~8 to ~13 FLOP/byte out of L2 and quadrupling WMMA
// density per wave. Per-lane fragments are contiguous b128 loads matching the
// documented CDNA5 16-bit A(16x32)/B(32x16) VGPR layouts.
// Optional outputs: f32 C, f16 row-major Ch, f16 per-batch-transposed ChT (V).
// Optional per-batch weight select (tfp_w[task_id]).
// ---------------------------------------------------------------------------
__global__ __launch_bounds__(32) void wmma_gemm(
    const _Float16* __restrict__ A, int lda,
    const _Float16* __restrict__ BwT,
    const float* __restrict__ bias,
    const float* __restrict__ residual, int ldr,
    float* __restrict__ C, int ldc,
    _Float16* __restrict__ Ch, int ldch,
    _Float16* __restrict__ ChT, int chTld, long chTbstride, int rowsPerBatchT,
    int Kd, int act,
    const int* __restrict__ wsel, int wselRows, long wStride, int biasStride) {
  int n0 = blockIdx.x * 64, m0 = blockIdx.y * 16;
  int lane = threadIdx.x, lm = lane & 15, hi = lane >> 4;
  const _Float16* Bp = BwT;
  const float* biasP = bias;
  if (wsel) {
    int t = wsel[m0 / wselRows];
    Bp += (long)t * wStride;
    if (biasP) biasP += (long)t * biasStride;
  }
  v8f acc[4] = {{}, {}, {}, {}};
  const _Float16* arow = A + (long)(m0 + lm) * lda;
  const _Float16* brow[4];
#pragma unroll
  for (int t = 0; t < 4; t++) brow[t] = Bp + (long)(n0 + t * 16 + lm) * Kd;

  for (int kk = 0; kk < Kd; kk += 32) {
    v8h a0 = *(const v8h*)(arow + kk + hi * 8);
    v8h a1 = *(const v8h*)(arow + kk + 16 + hi * 8);
    v16h af;
#pragma unroll
    for (int i = 0; i < 8; i++) { af[i] = a0[i]; af[i + 8] = a1[i]; }
#pragma unroll
    for (int t = 0; t < 4; t++) {
      v8h b0 = *(const v8h*)(brow[t] + kk + hi * 8);
      v8h b1 = *(const v8h*)(brow[t] + kk + 16 + hi * 8);
      v16h bf;
#pragma unroll
      for (int i = 0; i < 8; i++) { bf[i] = b0[i]; bf[i + 8] = b1[i]; }
      acc[t] = __builtin_amdgcn_wmma_f32_16x16x32_f16(false, af, false, bf, (short)0, acc[t], false, false);
    }
  }
#pragma unroll
  for (int t = 0; t < 4; t++) {
    int col = n0 + t * 16 + lm;
    float bvv = biasP ? biasP[col] : 0.0f;
#pragma unroll
    for (int r = 0; r < 8; r++) {
      int row = m0 + r + hi * 8;
      float v = acc[t][r] + bvv;
      if (residual) v += residual[(long)row * ldr + col];
      if (act) v = gelu_f(v);
      if (C)  C[(long)row * ldc + col] = v;
      if (Ch) Ch[(long)row * ldch + col] = (_Float16)v;
      if (ChT) {
        int bI = row / rowsPerBatchT;
        int nn = row - bI * rowsPerBatchT;
        ChT[(long)bI * chTbstride + (long)col * chTld + nn] = (_Float16)v;
      }
    }
  }
}

// ---------------------------------------------------------------------------
// Fused cross-attention: one wave per (16 queries, head, batch).
// scores = QK^T/sqrt(dh) + analytic distance bias (masked at n>=Ns), staged in
// LDS; row softmax; O = P@V via WMMA over transposed V.
// ---------------------------------------------------------------------------
__global__ __launch_bounds__(32) void attn_kernel(
    const _Float16* __restrict__ q16, const _Float16* __restrict__ k16,
    const _Float16* __restrict__ vT16,
    const float* __restrict__ meas_xy, const float* __restrict__ bs_xy,
    _Float16* __restrict__ o16) {
  __shared__ float sc[16 * kNsP];                 // 34,816 B
  int k0 = blockIdx.x * 16, h = blockIdx.y, b = blockIdx.z;
  int lane = threadIdx.x, lm = lane & 15, hi = lane >> 4;
  const float inv_sqrt = 0.17677669529663687f;   // 1/sqrt(32)

  // Q fragment (A-matrix 16x32 f16)
  const _Float16* qrow = q16 + ((long)(b * kK + k0 + lm)) * kD + h * kDH;
  v8h qa = *(const v8h*)(qrow + hi * 16);
  v8h qb = *(const v8h*)(qrow + hi * 16 + 8);
  v16h qf;
#pragma unroll
  for (int i = 0; i < 8; i++) { qf[i] = qa[i]; qf[i + 8] = qb[i]; }

  // scores
  for (int nt = 0; nt < kNsP / 16; nt++) {
    int n0 = nt * 16;
    const _Float16* krow = k16 + ((long)(b * kNsP + n0 + lm)) * kD + h * kDH + hi * 16;
    v8h kb0 = *(const v8h*)(krow);
    v8h kb1 = *(const v8h*)(krow + 8);
    v16h kf;
#pragma unroll
    for (int i = 0; i < 8; i++) { kf[i] = kb0[i]; kf[i + 8] = kb1[i]; }
    v8f z = {};
    v8f s = __builtin_amdgcn_wmma_f32_16x16x32_f16(false, qf, false, kf, (short)0, z, false, false);

    int n = n0 + lm;
    bool valid = (n < kNs);
    float sx = 0.0f, sy = 0.0f;
    if (n < kN)       { sx = meas_xy[((long)b * kN + n) * 2]; sy = meas_xy[((long)b * kN + n) * 2 + 1]; }
    else if (n == kN) { sx = bs_xy[b * 2]; sy = bs_xy[b * 2 + 1]; }
#pragma unroll
    for (int r = 0; r < 8; r++) {
      int kq = k0 + r + hi * 8;
      float gx = ((kq & 63) + 0.5f) * (1.0f / kG);
      float gy = ((kq >> 6) + 0.5f) * (1.0f / kG);
      float val;
      if (valid) {
        float dx = gx - sx, dy = gy - sy;
        val = s[r] * inv_sqrt - sqrtf(dx * dx + dy * dy);
      } else {
        val = -1e30f;
      }
      sc[(r + hi * 8) * kNsP + n] = val;
    }
  }
  __syncthreads();

  // softmax per row (lanes 0..15)
  if (lane < 16) {
    float mx = -1e30f;
    for (int n = 0; n < kNsP; n++) mx = fmaxf(mx, sc[lane * kNsP + n]);
    float sum = 0.0f;
    for (int n = 0; n < kNsP; n++) { float e = __expf(sc[lane * kNsP + n] - mx); sc[lane * kNsP + n] = e; sum += e; }
    float inv = 1.0f / sum;
    for (int n = 0; n < kNsP; n++) sc[lane * kNsP + n] *= inv;
  }
  __syncthreads();

  // O = P @ V (dh split into two 16-col tiles)
  v8f o0 = {}, o1 = {};
  for (int ct = 0; ct < kNsP / 32; ct++) {
    int nc = ct * 32;
    v16h pf;
#pragma unroll
    for (int i = 0; i < 8; i++) {
      pf[i]     = (_Float16)sc[lm * kNsP + nc + hi * 16 + i];
      pf[i + 8] = (_Float16)sc[lm * kNsP + nc + hi * 16 + 8 + i];
    }
    const _Float16* v0 = vT16 + ((long)b * kD + h * kDH + lm) * kNsP + nc + hi * 16;
    const _Float16* v1 = vT16 + ((long)b * kD + h * kDH + 16 + lm) * kNsP + nc + hi * 16;
    v8h va0 = *(const v8h*)(v0), va1 = *(const v8h*)(v0 + 8);
    v8h vb0 = *(const v8h*)(v1), vb1 = *(const v8h*)(v1 + 8);
    v16h vf0, vf1;
#pragma unroll
    for (int i = 0; i < 8; i++) { vf0[i] = va0[i]; vf0[i + 8] = va1[i]; vf1[i] = vb0[i]; vf1[i + 8] = vb1[i]; }
    o0 = __builtin_amdgcn_wmma_f32_16x16x32_f16(false, pf, false, vf0, (short)0, o0, false, false);
    o1 = __builtin_amdgcn_wmma_f32_16x16x32_f16(false, pf, false, vf1, (short)0, o1, false, false);
  }
#pragma unroll
  for (int r = 0; r < 8; r++) {
    long row = (long)(b * kK + k0 + r + hi * 8) * kD + h * kDH;
    o16[row + lm]      = (_Float16)o0[r];
    o16[row + 16 + lm] = (_Float16)o1[r];
  }
}

// ---------------------------------------------------------------------------
// head: out[row] = tf[row] @ head_w + head_b
// ---------------------------------------------------------------------------
__global__ void head_kernel(const float* __restrict__ tf, const float* __restrict__ hw,
                            const float* __restrict__ hb, float* __restrict__ out) {
  long row = (long)blockIdx.x * blockDim.x + threadIdx.x;
  if (row >= (long)kB * kK) return;
  float s = hb[0];
  const float* t = tf + row * kD;
  for (int d = 0; d < kD; d++) s += t[d] * hw[d];
  out[row] = s;
}

// ---------------------------------------------------------------------------
// host side
// ---------------------------------------------------------------------------
static void gemm(hipStream_t s, const _Float16* A, int lda, const _Float16* BwT,
                 const float* bias, const float* res, int ldr,
                 float* C, int ldc, _Float16* Ch, int ldch,
                 _Float16* ChT, int chTld, long chTb, int rowsT,
                 int M, int N, int Kd, int act,
                 const int* wsel = nullptr, int wselRows = 1, long wStride = 0, int biasStride = 0) {
  dim3 grd(N / 64, M / 16);
  wmma_gemm<<<grd, 32, 0, s>>>(A, lda, BwT, bias, res, ldr, C, ldc, Ch, ldch,
                               ChT, chTld, chTb, rowsT, Kd, act, wsel, wselRows, wStride, biasStride);
}

static inline int nblk(long n) { return (int)((n + 255) / 256); }

extern "C" void kernel_launch(void* const* d_in, const int* in_sizes, int n_in,
                              void* d_out, int out_size, void* d_ws, size_t ws_size,
                              hipStream_t stream) {
  (void)in_sizes; (void)n_in; (void)out_size; (void)ws_size;
  const float* meas_xy   = (const float*)d_in[0];
  const float* meas_v    = (const float*)d_in[1];
  const float* bs_xy     = (const float*)d_in[2];
  const int*   task_id   = (const int*)d_in[3];
  const float* city      = (const float*)d_in[4];
  const float* mp_w1     = (const float*)d_in[5];
  const float* mp_b1     = (const float*)d_in[6];
  const float* mp_w2     = (const float*)d_in[7];
  const float* mp_b2     = (const float*)d_in[8];
  const float* bp_w1     = (const float*)d_in[9];
  const float* bp_b1     = (const float*)d_in[10];
  const float* bp_w2     = (const float*)d_in[11];
  const float* bp_b2     = (const float*)d_in[12];
  const float* env_w1    = (const float*)d_in[13];
  const float* env_b1    = (const float*)d_in[14];
  const float* env_w2    = (const float*)d_in[15];
  const float* env_b2    = (const float*)d_in[16];
  const float* env_w3    = (const float*)d_in[17];
  const float* env_b3    = (const float*)d_in[18];
  const float* task_emb  = (const float*)d_in[19];
  const float* tp_w      = (const float*)d_in[20];
  const float* tp_b      = (const float*)d_in[21];
  const float* grid_pos  = (const float*)d_in[22];
  const float* bdw       = (const float*)d_in[23];
  const float* bdb       = (const float*)d_in[24];
  const float* ln1w      = (const float*)d_in[25];
  const float* ln1b      = (const float*)d_in[26];
  const float* wq        = (const float*)d_in[27];
  const float* bq        = (const float*)d_in[28];
  const float* wk        = (const float*)d_in[29];
  const float* bk        = (const float*)d_in[30];
  const float* wv        = (const float*)d_in[31];
  const float* bv        = (const float*)d_in[32];
  const float* wo        = (const float*)d_in[33];
  const float* bo        = (const float*)d_in[34];
  const float* ln2w      = (const float*)d_in[35];
  const float* ln2b      = (const float*)d_in[36];
  const float* ffw1      = (const float*)d_in[37];
  const float* ffb1      = (const float*)d_in[38];
  const float* ffw2      = (const float*)d_in[39];
  const float* ffb2      = (const float*)d_in[40];
  const float* tfpw      = (const float*)d_in[41];
  const float* tfpb      = (const float*)d_in[42];
  const float* head_w    = (const float*)d_in[43];
  const float* head_b    = (const float*)d_in[44];
  float* out = (float*)d_out;

  char* wsb = (char*)d_ws;
  size_t off = 0;
  auto alloc = [&](size_t bytes) -> char* {
    off = (off + 255) & ~(size_t)255;
    char* p = wsb + off;
    off += bytes;
    return p;
  };
  // f16 activations
  _Float16* h1    = (_Float16*)alloc((size_t)kB * kNsP * kD * 2);
  _Float16* src16 = (_Float16*)alloc((size_t)kB * kNsP * kD * 2);
  _Float16* q16   = (_Float16*)alloc((size_t)kB * kK * kD * 2);
  _Float16* k16   = (_Float16*)alloc((size_t)kB * kNsP * kD * 2);
  _Float16* vT16  = (_Float16*)alloc((size_t)kB * kD * kNsP * 2);
  _Float16* o16   = (_Float16*)alloc((size_t)kB * kK * kD * 2);
  _Float16* hn16  = (_Float16*)alloc((size_t)kB * kK * kD * 2);
  _Float16* ff16  = (_Float16*)alloc((size_t)kB * kK * kDFF * 2);
  _Float16* gh16  = (_Float16*)alloc((size_t)kB * kK * kD * 2);
  // f32 buffers
  float* gr0  = (float*)alloc((size_t)kB * kK * kD * 4);
  float* gr1  = (float*)alloc((size_t)kB * kK * kD * 4);
  float* gr2  = (float*)alloc((size_t)kB * kK * kD * 4);
  float* tfb  = (float*)alloc((size_t)kB * kK * kD * 4);
  float* dens = (float*)alloc((size_t)kB * kK * 4);
  float* tvec = (float*)alloc((size_t)kB * kD * 4);
  float* c1   = (float*)alloc((size_t)kB * kEC * kG * kG * 4);
  float* c2   = (float*)alloc((size_t)kB * kEC * kG * kG * 4);
  // f16 transposed weights
  _Float16* mpw2T = (_Float16*)alloc((size_t)kD * kD * 2);
  _Float16* wqT   = (_Float16*)alloc((size_t)2 * kD * kD * 2);
  _Float16* wkT   = (_Float16*)alloc((size_t)2 * kD * kD * 2);
  _Float16* wvT   = (_Float16*)alloc((size_t)2 * kD * kD * 2);
  _Float16* woT   = (_Float16*)alloc((size_t)2 * kD * kD * 2);
  _Float16* ffw1T = (_Float16*)alloc((size_t)2 * kD * kDFF * 2);
  _Float16* ffw2T = (_Float16*)alloc((size_t)2 * kDFF * kD * 2);
  _Float16* tfpT  = (_Float16*)alloc((size_t)kT * kD * kD * 2);

  // --- weight conversions (f32 -> f16 transposed) ---
  cvt_transpose_f16<<<nblk((long)kD * kD), 256, 0, stream>>>(mp_w2, mpw2T, kD, kD, 1);
  cvt_transpose_f16<<<nblk(2L * kD * kD), 256, 0, stream>>>(wq, wqT, kD, kD, 2);
  cvt_transpose_f16<<<nblk(2L * kD * kD), 256, 0, stream>>>(wk, wkT, kD, kD, 2);
  cvt_transpose_f16<<<nblk(2L * kD * kD), 256, 0, stream>>>(wv, wvT, kD, kD, 2);
  cvt_transpose_f16<<<nblk(2L * kD * kD), 256, 0, stream>>>(wo, woT, kD, kD, 2);
  cvt_transpose_f16<<<nblk(2L * kD * kDFF), 256, 0, stream>>>(ffw1, ffw1T, kD, kDFF, 2);
  cvt_transpose_f16<<<nblk(2L * kDFF * kD), 256, 0, stream>>>(ffw2, ffw2T, kDFF, kD, 2);
  cvt_transpose_f16<<<nblk((long)kT * kD * kD), 256, 0, stream>>>(tfpw, tfpT, kD, kD, kT);

  // --- tokens ---
  h1_kernel<<<nblk((long)kB * kNsP * kD), 256, 0, stream>>>(meas_xy, meas_v, mp_w1, mp_b1, h1);
  gemm(stream, h1, kD, mpw2T, mp_b2, nullptr, 0, nullptr, 0, src16, kD,
       nullptr, 0, 0, 1, kB * kNsP, kD, kD, 0);
  bs_kernel<<<kB, 256, 0, stream>>>(bs_xy, bp_w1, bp_b1, bp_w2, bp_b2, src16);

  // --- task vec, density, env CNN, grid assembly ---
  taskvec_kernel<<<kB, 256, 0, stream>>>(task_id, task_emb, tp_w, tp_b, tvec);
  density_kernel<<<dim3(kK / 256, kB), 256, 0, stream>>>(meas_xy, dens);
  conv1_k<<<nblk((long)kB * kEC * kG * kG), 256, 0, stream>>>(city, env_w1, env_b1, c1);
  conv2_k<<<nblk((long)kB * kEC * kG * kG), 256, 0, stream>>>(c1, env_w2, env_b2, c2);
  conv3_assemble<<<nblk((long)kB * kK * kD), 256, 0, stream>>>(c2, env_w3, env_b3, grid_pos, tvec, gr0);

  // --- transformer layers ---
  for (int l = 0; l < 2; l++) {
    densadd_k<<<nblk((long)kB * kK * kD), 256, 0, stream>>>(gr0, dens, bdw + l * kD, bdb + l * kD, gr1);
    ln_kernel<<<kB * kK, 256, 0, stream>>>(gr1, hn16, ln1w + l * kD, ln1b + l * kD);
    gemm(stream, hn16, kD, wqT + (long)l * kD * kD, bq + l * kD, nullptr, 0,
         nullptr, 0, q16, kD, nullptr, 0, 0, 1, kB * kK, kD, kD, 0);
    gemm(stream, src16, kD, wkT + (long)l * kD * kD, bk + l * kD, nullptr, 0,
         nullptr, 0, k16, kD, nullptr, 0, 0, 1, kB * kNsP, kD, kD, 0);
    gemm(stream, src16, kD, wvT + (long)l * kD * kD, bv + l * kD, nullptr, 0,
         nullptr, 0, nullptr, 0, vT16, kNsP, (long)kD * kNsP, kNsP, kB * kNsP, kD, kD, 0);
    attn_kernel<<<dim3(kK / 16, kNH, kB), 32, 0, stream>>>(q16, k16, vT16, meas_xy, bs_xy, o16);
    gemm(stream, o16, kD, woT + (long)l * kD * kD, bo + l * kD, gr1, kD,
         gr2, kD, nullptr, 0, nullptr, 0, 0, 1, kB * kK, kD, kD, 0);
    ln_kernel<<<kB * kK, 256, 0, stream>>>(gr2, hn16, ln2w + l * kD, ln2b + l * kD);
    gemm(stream, hn16, kD, ffw1T + (long)l * kDFF * kD, ffb1 + l * kDFF, nullptr, 0,
         nullptr, 0, ff16, kDFF, nullptr, 0, 0, 1, kB * kK, kDFF, kD, 1);
    gemm(stream, ff16, kDFF, ffw2T + (long)l * kD * kDFF, ffb2 + l * kD, gr2, kD,
         gr0, kD, gh16, kD, nullptr, 0, 0, 1, kB * kK, kD, kDFF, 0);
  }

  // --- task-selected projection + head ---
  gemm(stream, gh16, kD, tfpT, tfpb, nullptr, 0, tfb, kD, nullptr, 0,
       nullptr, 0, 0, 1, kB * kK, kD, kD, 1, task_id, kK, (long)kD * kD, kD);
  head_kernel<<<nblk((long)kB * kK), 256, 0, stream>>>(tfb, head_w, head_b, out);
}
